// MultiHeadAttention_30279519437031
// MI455X (gfx1250) — matrix operand order repaired and checked by reading, not compiled, not run on hardware
//
#include <hip/hip_runtime.h>
#include <hip/hip_bf16.h>

typedef __attribute__((ext_vector_type(16))) _Float16 v16h;
typedef __attribute__((ext_vector_type(8)))  _Float16 v8h;
typedef __attribute__((ext_vector_type(8)))  float    v8f;

#define BATCH   2
#define SEQ     2048
#define DMODEL  768
#define NHEADS  12
#define DHEAD   64
#define NTOK    (BATCH * SEQ)       /* 4096 tokens */
#define KT_DM   (DMODEL / 32)       /* 24 K-steps over d_model (even) */

// ---------------- workspace layout (units: _Float16 elements) ----------------
#define XH_OFF   ((size_t)0)
#define WQP_OFF  (XH_OFF  + (size_t)NTOK * DMODEL)
#define WKP_OFF  (WQP_OFF + (size_t)NHEADS * DMODEL * DHEAD)
#define WVP_OFF  (WKP_OFF + (size_t)NHEADS * DMODEL * DHEAD)
#define WOP_OFF  (WVP_OFF + (size_t)NHEADS * DMODEL * DHEAD)
#define QH_OFF   (WOP_OFF + (size_t)DMODEL * DMODEL)
#define KH_OFF   (QH_OFF  + (size_t)BATCH * NHEADS * SEQ * DHEAD)
#define VT_OFF   (KH_OFF  + (size_t)BATCH * NHEADS * SEQ * DHEAD)
#define ZH_OFF   (VT_OFF  + (size_t)BATCH * NHEADS * SEQ * DHEAD)

// ---------------- WMMA helpers ----------------
__device__ __forceinline__ v8f wmma16(v16h a, v16h b, v8f c) {
  // D = A(16x32 f16) * B(32x16 f16) + C(16x16 f32)
  return __builtin_amdgcn_wmma_f32_16x16x32_f16(
      /*neg_a=*/false, a, /*neg_b=*/false, b,
      /*c_mod=*/(short)0, c, /*reuse_a=*/false, /*reuse_b=*/false);
}

// Load a 16x32 f16 A-fragment from a row-major matrix.
// A layout: lane L holds row M = L%16; halves 0..15 -> K = 16*(L/16)+i.
// Two contiguous b128 loads/lane.
__device__ __forceinline__ v16h load_a16(const _Float16* base, int stride, int lane) {
  const _Float16* p = base + (size_t)(lane & 15) * stride + ((lane >> 4) * 16);
  union { v16h v; v8h h2[2]; } u;
  u.h2[0] = *reinterpret_cast<const v8h*>(p);
  u.h2[1] = *reinterpret_cast<const v8h*>(p + 8);
  return u.v;
}

// ---------------- pack / convert kernels ----------------
__global__ void cvt_f32_to_f16_kernel(const float* __restrict__ src,
                                      _Float16* __restrict__ dst, int n) {
  int i = blockIdx.x * blockDim.x + threadIdx.x;
  if (i < n) dst[i] = (_Float16)src[i];
}

// Pack a batch of K x N row-major fp32 matrices into WMMA B-fragment order:
// dst[((mat*fpm + nt*ktC + kt) * 32 + lane) * 16 + i], where for lane L,
// half i corresponds to (K = kt*32 + (L/16)*16 + i, N = nt*16 + L%16).
__global__ void packB_kernel(const float* __restrict__ src, _Float16* __restrict__ dst,
                             int K, int N, int nMat) {
  const int tid  = blockIdx.x * blockDim.x + threadIdx.x;
  const int lane = tid & 31;
  int frag = tid >> 5;
  const int ktC = K >> 5, ntC = N >> 4;
  const int fpm = ktC * ntC;
  const int mat = frag / fpm;
  if (mat >= nMat) return;
  frag -= mat * fpm;
  const int nt = frag / ktC;
  const int kt = frag - nt * ktC;
  const float* s = src + (size_t)mat * K * N;
  _Float16*    d = dst + (((size_t)mat * fpm + frag) * 32 + lane) * 16;
  const int n  = nt * 16 + (lane & 15);
  const int kb = kt * 32 + ((lane >> 4) * 16);
#pragma unroll
  for (int i = 0; i < 16; ++i)
    d[i] = (_Float16)s[(size_t)(kb + i) * N + n];
}

// ---------------- QKV projection GEMM (fully-unrolled ping-pong) -----------
// grid = (NTOK/64) * NHEADS * 3, block = 128 (4 waves). Wave: 16 rows x 64 cols.
__global__ __launch_bounds__(128) void qkv_kernel(
    const _Float16* __restrict__ Xh, const _Float16* __restrict__ Wp,
    const float* __restrict__ bQ, const float* __restrict__ bK,
    const float* __restrict__ bV,
    _Float16* __restrict__ Qh, _Float16* __restrict__ Kh, _Float16* __restrict__ Vt) {
  const int bid     = blockIdx.x;
  const int rowTile = bid & 63;
  const int h       = (bid >> 6) % NHEADS;
  const int which   = bid / (64 * NHEADS);    // 0=Q 1=K 2=V
  const int w    = threadIdx.x >> 5;
  const int lane = threadIdx.x & 31;
  const int t0   = rowTile * 64 + w * 16;
  const int col  = lane & 15;
  const int rhi  = (lane >> 4) * 8;

  const _Float16* wf = Wp + ((size_t)which * NHEADS + h) * ((size_t)DMODEL * DHEAD);
  const float* bias  = (which == 0 ? bQ : (which == 1 ? bK : bV)) + h * DHEAD;

  const v8f vz = {0.f, 0.f, 0.f, 0.f, 0.f, 0.f, 0.f, 0.f};
  v8f acc[4];
#pragma unroll
  for (int n = 0; n < 4; ++n) acc[n] = vz;

  const _Float16* abase = Xh + (size_t)t0 * DMODEL;

  // ping-pong register sets A/B; fully unrolled -> no backedge, no copies
  v16h aA = load_a16(abase, DMODEL, lane);
  v16h bA[4];
#pragma unroll
  for (int n = 0; n < 4; ++n)
    bA[n] = *reinterpret_cast<const v16h*>(
        wf + ((size_t)(n * KT_DM) * 32 + lane) * 16);

#pragma unroll
  for (int kt = 0; kt < KT_DM; kt += 2) {
    // issue B-set loads (kt+1), then consume A-set (kt)
    v16h aB = load_a16(abase + (kt + 1) * 32, DMODEL, lane);
    v16h bB[4];
#pragma unroll
    for (int n = 0; n < 4; ++n)
      bB[n] = *reinterpret_cast<const v16h*>(
          wf + ((size_t)(n * KT_DM + kt + 1) * 32 + lane) * 16);
#pragma unroll
    for (int n = 0; n < 4; ++n)
      acc[n] = wmma16(aA, bA[n], acc[n]);

    // issue A-set loads (kt+2; none on final stage), then consume B-set (kt+1)
    if (kt + 2 < KT_DM) {
      aA = load_a16(abase + (kt + 2) * 32, DMODEL, lane);
#pragma unroll
      for (int n = 0; n < 4; ++n)
        bA[n] = *reinterpret_cast<const v16h*>(
            wf + ((size_t)(n * KT_DM + kt + 2) * 32 + lane) * 16);
    }
#pragma unroll
    for (int n = 0; n < 4; ++n)
      acc[n] = wmma16(aB, bB[n], acc[n]);
  }

#pragma unroll
  for (int n = 0; n < 4; ++n) {
    const int d  = n * 16 + col;
    const float bv = bias[d];
#pragma unroll
    for (int r = 0; r < 8; ++r) {
      const int t  = t0 + rhi + r;
      const int bb = t >> 11;            // token -> batch
      const int s  = t & (SEQ - 1);      // token -> seq position
      const float v = acc[n][r] + bv;
      if (which == 2) {                  // V stored transposed: [b,h,d,s]
        Vt[(((size_t)bb * NHEADS + h) * DHEAD + d) * SEQ + s] = (_Float16)v;
      } else {                           // Q,K stored [b,h,s,d]
        _Float16* dst = which ? Kh : Qh;
        dst[(((size_t)bb * NHEADS + h) * SEQ + s) * DHEAD + d] = (_Float16)v;
      }
    }
  }
}

// ---------------- flash attention (causal, online softmax) ----------------
// grid = BATCH * NHEADS * (SEQ/64), block = 128. One wave owns 16 q-rows.
__global__ __launch_bounds__(128) void attn_kernel(
    const _Float16* __restrict__ Qh, const _Float16* __restrict__ Kh,
    const _Float16* __restrict__ Vt, _Float16* __restrict__ Zh) {
  __shared__ __attribute__((aligned(32))) _Float16 pbuf[4 * 16 * 32];

  const int bid  = blockIdx.x;
  const int qt   = bid & 31;                 // SEQ/64 = 32 q-tiles
  const int h    = (bid >> 5) % NHEADS;
  const int b    = bid / (32 * NHEADS);
  const int w    = threadIdx.x >> 5;
  const int lane = threadIdx.x & 31;
  const int q0   = qt * 64 + w * 16;
  const int col  = lane & 15;
  const int rhi  = (lane >> 4) * 8;
  const int hi16 = (lane >> 4) * 16;

  const _Float16* Qb = Qh + (((size_t)b * NHEADS + h) * SEQ) * DHEAD;
  const _Float16* Kb = Kh + (((size_t)b * NHEADS + h) * SEQ) * DHEAD;
  const _Float16* Vb = Vt + (((size_t)b * NHEADS + h) * DHEAD) * SEQ;
  _Float16* myp = pbuf + w * (16 * 32);

  // Q fragments for this wave's 16 rows (d 0..31 and 32..63)
  const v16h qa0 = load_a16(Qb + (size_t)q0 * DHEAD + 0,  DHEAD, lane);
  const v16h qa1 = load_a16(Qb + (size_t)q0 * DHEAD + 32, DHEAD, lane);

  const v8f vz = {0.f, 0.f, 0.f, 0.f, 0.f, 0.f, 0.f, 0.f};
  float m[8], l[8];
  v8f o[4];
#pragma unroll
  for (int r = 0; r < 8; ++r) { m[r] = -1e30f; l[r] = 0.f; }
#pragma unroll
  for (int n = 0; n < 4; ++n) o[n] = vz;

  const int nchunks = (q0 + 16 + 31) >> 5;   // causal bound, wave-uniform
  for (int c = 0; c < nchunks; ++c) {
    const int kb = c * 32;

    // ---- issue ALL chunk loads up front: K fragments feed the score WMMAs,
    //      V fragments are consumed only after softmax (latency fully hidden)
    const _Float16* kp0 = Kb + (size_t)(kb + col) * DHEAD + hi16;
    const _Float16* kp1 = Kb + (size_t)(kb + 16 + col) * DHEAD + hi16;
    const v16h kf00 = *reinterpret_cast<const v16h*>(kp0);
    const v16h kf10 = *reinterpret_cast<const v16h*>(kp0 + 32);
    const v16h kf01 = *reinterpret_cast<const v16h*>(kp1);
    const v16h kf11 = *reinterpret_cast<const v16h*>(kp1 + 32);
    v16h vf[4];
#pragma unroll
    for (int n = 0; n < 4; ++n)
      vf[n] = *reinterpret_cast<const v16h*>(
          Vb + (size_t)(n * 16 + col) * SEQ + kb + hi16);
    if (c + 1 < nchunks) {                   // warm WGP$/L2 for next chunk
      __builtin_prefetch(Kb + (size_t)(kb + 32 + col) * DHEAD, 0, 2);
      __builtin_prefetch(Vb + (size_t)col * SEQ + kb + 32, 0, 2);
    }

    // ---- scores: S = Q * K^T, two 16x16 tiles
    v8f s0 = vz, s1 = vz;
    s0 = wmma16(qa0, kf00, s0);
    s0 = wmma16(qa1, kf10, s0);
    s1 = wmma16(qa0, kf01, s1);
    s1 = wmma16(qa1, kf11, s1);

    // ---- scale, causal mask, online softmax (row = 16 lanes of a half-wave)
    float alpha[8], p0a[8], p1a[8];
#pragma unroll
    for (int r = 0; r < 8; ++r) {
      const int q = q0 + rhi + r;
      float v0 = s0[r] * 0.125f; if (kb + col > q)      v0 = -1e30f;
      float v1 = s1[r] * 0.125f; if (kb + 16 + col > q) v1 = -1e30f;
      float mx = fmaxf(v0, v1);
      mx = fmaxf(mx, __shfl_xor(mx, 1, 32));
      mx = fmaxf(mx, __shfl_xor(mx, 2, 32));
      mx = fmaxf(mx, __shfl_xor(mx, 4, 32));
      mx = fmaxf(mx, __shfl_xor(mx, 8, 32));
      const float mn = fmaxf(m[r], mx);
      alpha[r] = __expf(m[r] - mn);
      m[r] = mn;
      const float p0 = __expf(v0 - mn);
      const float p1 = __expf(v1 - mn);
      float rs = p0 + p1;
      rs += __shfl_xor(rs, 1, 32);
      rs += __shfl_xor(rs, 2, 32);
      rs += __shfl_xor(rs, 4, 32);
      rs += __shfl_xor(rs, 8, 32);
      l[r] = l[r] * alpha[r] + rs;
      p0a[r] = p0; p1a[r] = p1;
    }
#pragma unroll
    for (int n = 0; n < 4; ++n)
#pragma unroll
      for (int r = 0; r < 8; ++r) o[n][r] = o[n][r] * alpha[r];

    // ---- C-layout -> A-fragment via per-wave LDS (16x32 f16 tile)
#pragma unroll
    for (int r = 0; r < 8; ++r) {
      const int row = rhi + r;
      myp[row * 32 + col]      = (_Float16)p0a[r];
      myp[row * 32 + 16 + col] = (_Float16)p1a[r];
    }
    asm volatile("s_wait_dscnt 0" ::: "memory");   // wave-local LDS RAW
    v16h pa;
    {
      const _Float16* p = myp + (lane & 15) * 32 + ((lane >> 4) * 16);
      union { v16h v; v8h h2[2]; } u;
      u.h2[0] = *reinterpret_cast<const v8h*>(p);
      u.h2[1] = *reinterpret_cast<const v8h*>(p + 8);
      pa = u.v;
    }
    // ---- O += P * V  (V fragments already resident in VGPRs)
#pragma unroll
    for (int n = 0; n < 4; ++n)
      o[n] = wmma16(pa, vf[n], o[n]);
  }

  // ---- normalize and store Z as f16 [b, s, h*64+d]
  float inv[8];
#pragma unroll
  for (int r = 0; r < 8; ++r) inv[r] = 1.0f / l[r];
#pragma unroll
  for (int n = 0; n < 4; ++n)
#pragma unroll
    for (int r = 0; r < 8; ++r) {
      const int q = q0 + rhi + r;
      const int d = n * 16 + col;
      Zh[((size_t)(b * SEQ + q)) * DMODEL + h * DHEAD + d] =
          (_Float16)(o[n][r] * inv[r]);
    }
}

// ---------------- output projection GEMM (fully-unrolled ping-pong) --------
// grid = (NTOK/64) * (DMODEL/64), block = 128.
__global__ __launch_bounds__(128) void oproj_kernel(
    const _Float16* __restrict__ Zh, const _Float16* __restrict__ Wop,
    const float* __restrict__ bO, float* __restrict__ out) {
  const int bid     = blockIdx.x;
  const int rowTile = bid & 63;
  const int colTile = bid >> 6;              // 0..11
  const int w    = threadIdx.x >> 5;
  const int lane = threadIdx.x & 31;
  const int t0   = rowTile * 64 + w * 16;
  const int col  = lane & 15;
  const int rhi  = (lane >> 4) * 8;

  const v8f vz = {0.f, 0.f, 0.f, 0.f, 0.f, 0.f, 0.f, 0.f};
  v8f acc[4];
#pragma unroll
  for (int n = 0; n < 4; ++n) acc[n] = vz;

  const _Float16* abase = Zh + (size_t)t0 * DMODEL;
  const _Float16* wbase = Wop + ((size_t)colTile * 4 * KT_DM) * 32 * 16;

  v16h aA = load_a16(abase, DMODEL, lane);
  v16h bA[4];
#pragma unroll
  for (int n = 0; n < 4; ++n)
    bA[n] = *reinterpret_cast<const v16h*>(
        wbase + ((size_t)(n * KT_DM) * 32 + lane) * 16);

#pragma unroll
  for (int kt = 0; kt < KT_DM; kt += 2) {
    v16h aB = load_a16(abase + (kt + 1) * 32, DMODEL, lane);
    v16h bB[4];
#pragma unroll
    for (int n = 0; n < 4; ++n)
      bB[n] = *reinterpret_cast<const v16h*>(
          wbase + ((size_t)(n * KT_DM + kt + 1) * 32 + lane) * 16);
#pragma unroll
    for (int n = 0; n < 4; ++n)
      acc[n] = wmma16(aA, bA[n], acc[n]);

    if (kt + 2 < KT_DM) {
      aA = load_a16(abase + (kt + 2) * 32, DMODEL, lane);
#pragma unroll
      for (int n = 0; n < 4; ++n)
        bA[n] = *reinterpret_cast<const v16h*>(
            wbase + ((size_t)(n * KT_DM + kt + 2) * 32 + lane) * 16);
    }
#pragma unroll
    for (int n = 0; n < 4; ++n)
      acc[n] = wmma16(aB, bB[n], acc[n]);
  }

#pragma unroll
  for (int n = 0; n < 4; ++n) {
    const int cg = colTile * 64 + n * 16 + col;
    const float bv = bO[cg];
#pragma unroll
    for (int r = 0; r < 8; ++r)
      out[(size_t)(t0 + rhi + r) * DMODEL + cg] = acc[n][r] + bv;
  }
}

// ---------------- host launcher ----------------
extern "C" void kernel_launch(void* const* d_in, const int* in_sizes, int n_in,
                              void* d_out, int out_size, void* d_ws, size_t ws_size,
                              hipStream_t stream) {
  const float* x  = (const float*)d_in[0];
  const float* WQ = (const float*)d_in[1];
  const float* bQ = (const float*)d_in[2];
  const float* WK = (const float*)d_in[3];
  const float* bK = (const float*)d_in[4];
  const float* WV = (const float*)d_in[5];
  const float* bV = (const float*)d_in[6];
  const float* WO = (const float*)d_in[7];
  const float* bO = (const float*)d_in[8];
  float* out = (float*)d_out;
  _Float16* ws = (_Float16*)d_ws;

  _Float16* Xh  = ws + XH_OFF;
  _Float16* WQp = ws + WQP_OFF;
  _Float16* WKp = ws + WKP_OFF;
  _Float16* WVp = ws + WVP_OFF;
  _Float16* WOp = ws + WOP_OFF;
  _Float16* Qh  = ws + QH_OFF;
  _Float16* Kh  = ws + KH_OFF;
  _Float16* Vt  = ws + VT_OFF;
  _Float16* Zh  = ws + ZH_OFF;

  // 1) convert x to f16, pack all weights into WMMA B-fragment order
  cvt_f32_to_f16_kernel<<<(NTOK * DMODEL + 255) / 256, 256, 0, stream>>>(
      x, Xh, NTOK * DMODEL);
  const int wThreads = NHEADS * (DMODEL / 32) * (DHEAD / 16) * 32;   // 36864
  packB_kernel<<<(wThreads + 255) / 256, 256, 0, stream>>>(WQ, WQp, DMODEL, DHEAD, NHEADS);
  packB_kernel<<<(wThreads + 255) / 256, 256, 0, stream>>>(WK, WKp, DMODEL, DHEAD, NHEADS);
  packB_kernel<<<(wThreads + 255) / 256, 256, 0, stream>>>(WV, WVp, DMODEL, DHEAD, NHEADS);
  const int oThreads = (DMODEL / 32) * (DMODEL / 16) * 32;           // 36864
  packB_kernel<<<(oThreads + 255) / 256, 256, 0, stream>>>(WO, WOp, DMODEL, DMODEL, 1);

  // 2) QKV projections (WMMA), 3) flash attention (WMMA), 4) output proj (WMMA)
  qkv_kernel<<<(NTOK / 64) * NHEADS * 3, 128, 0, stream>>>(
      Xh, WQp, bQ, bK, bV, Qh, Kh, Vt);
  attn_kernel<<<BATCH * NHEADS * (SEQ / 64), 128, 0, stream>>>(Qh, Kh, Vt, Zh);
  oproj_kernel<<<(NTOK / 64) * (DMODEL / 64), 128, 0, stream>>>(Zh, WOp, bO, out);

  (void)in_sizes; (void)n_in; (void)out_size; (void)ws_size;
}
